// AMGModel_49254684951072
// MI455X (gfx1250) — matrix operand order repaired and checked by reading, not compiled, not run on hardware
//
#include <hip/hip_runtime.h>
#include <hip/hip_bf16.h>

// ---------------------------------------------------------------------------
// CDNA5 (gfx1250) implementation of the AMG GNN model.
//   * All dense layers run on V_WMMA_F32_16X16X32_BF16 (bf16 in, f32 acc).
//   * Weights pre-packed to transposed bf16 so B-fragments are contiguous.
//   * Activation tiles staged into LDS with GLOBAL_LOAD_ASYNC_TO_LDS_B128
//     (ASYNCcnt) where no on-the-fly conversion is needed.
//   * SAGE aggregation via global_atomic_add_f32 into an L2-resident buffer.
// ---------------------------------------------------------------------------

typedef __bf16 bf16_t;
typedef bf16_t v16bf __attribute__((ext_vector_type(16)));
typedef float  v8f   __attribute__((ext_vector_type(8)));

union FragU { v16bf v; unsigned int u[8]; };

// Same-wave LDS ordering fence (buffers are wave-private; no block barrier).
__device__ __forceinline__ void wave_lds_fence() {
  asm volatile("s_wait_dscnt 0" ::: "memory");
}

// Wait for all outstanding async (global->LDS) transfers of this wave.
__device__ __forceinline__ void async_fence() {
  asm volatile("s_wait_asynccnt 0x0" ::: "memory");
}

// One 16-byte chunk: global memory -> LDS, bypassing VGPRs (ASYNCcnt).
// lds_addr = low 32 bits of the generic pointer (ISA 10.2: LDS aperture).
__device__ __forceinline__ void async_load_b128(const bf16_t* lds_dst, const void* gsrc) {
  const unsigned lds_addr = (unsigned)(unsigned long long)lds_dst;
  asm volatile("global_load_async_to_lds_b128 %0, %1, off"
               :: "v"(lds_addr), "v"(gsrc) : "memory");
}

// ---------------------------------------------------------------------------
// One 16x16 output tile: acc += A[16xKDIM] (LDS, row-major, stride ldi)
//                             @ Wt[NOUT x KDIM] (global, row n = output col).
// Fragment layouts per CDNA5 ISA 7.12.2 (16-bit A 16x32; B 32x16).
// ---------------------------------------------------------------------------
template<int KDIM>
__device__ __forceinline__ v8f wmma_rowtile(const bf16_t* __restrict__ actL, int ldi,
                                            const bf16_t* __restrict__ Wt, int n0,
                                            int lane) {
  v8f acc = {0.f, 0.f, 0.f, 0.f, 0.f, 0.f, 0.f, 0.f};
  const int half = lane >> 4;
  const int rA   = lane & 15;
  const int colB = n0 + (lane & 15);
  #pragma unroll
  for (int k0 = 0; k0 < KDIM; k0 += 32) {
    FragU a, b;
    // A 16x32 bf16: lanes0-15 hold K=k0..k0+7 (v0..3) and k0+16..k0+23 (v4..7);
    // lanes16-31 the +8 halves.
    const bf16_t* ap = actL + rA * ldi + k0 + half * 8;
    #pragma unroll
    for (int r = 0; r < 4; ++r) {
      a.u[r]     = *(const unsigned int*)(ap + 2 * r);
      a.u[4 + r] = *(const unsigned int*)(ap + 16 + 2 * r);
    }
    // B 32x16 bf16: lane = column; lanes0-15 K=k0..k0+7 and k0+16..k0+23, lanes16-31 the +8 halves.
    const bf16_t* bp = Wt + (size_t)colB * KDIM + k0 + half * 8;
    #pragma unroll
    for (int r = 0; r < 8; ++r)
      b.u[r] = *(const unsigned int*)(bp + 2 * r + (r >> 2) * 8);
    acc = __builtin_amdgcn_wmma_f32_16x16x32_bf16(false, a.v, false, b.v,
                                                  (short)0, acc, false, false);
  }
  return acc;
}

// C/D f32 16x16 layout: lane = column; lanes0-15 rows 0..7 (v0..7), lanes16-31 rows 8..15.
__device__ __forceinline__ void store_tile_lds(v8f c, const float* __restrict__ bias,
                                               int n0, bool relu,
                                               bf16_t* __restrict__ outL, int ldo, int lane) {
  const int col = lane & 15;
  const int rb  = (lane >> 4) * 8;
  const float bv = bias[n0 + col];
  #pragma unroll
  for (int m = 0; m < 8; ++m) {
    float f = c[m] + bv;
    if (relu) f = fmaxf(f, 0.f);
    outL[(rb + m) * ldo + n0 + col] = (bf16_t)f;
  }
}

__device__ __forceinline__ void store_tile_glb(v8f c, const float* __restrict__ bias,
                                               int n0, bool relu,
                                               bf16_t* __restrict__ g, int ldg, int lane) {
  const int col = lane & 15;
  const int rb  = (lane >> 4) * 8;
  const float bv = bias[n0 + col];
  #pragma unroll
  for (int m = 0; m < 8; ++m) {
    float f = c[m] + bv;
    if (relu) f = fmaxf(f, 0.f);
    g[(size_t)(rb + m) * ldg + n0 + col] = (bf16_t)f;
  }
}

// ---------------------------------------------------------------------------
// Weight prep: f32 [K,N] row-major -> bf16 [N, ldt] transposed at col offset.
// ---------------------------------------------------------------------------
__global__ __launch_bounds__(256)
void transpose_to_bf16_kernel(const float* __restrict__ W, bf16_t* __restrict__ Wt,
                              int K, int N, int ldt, int kofs) {
  const int i = blockIdx.x * 256 + threadIdx.x;
  if (i >= K * N) return;
  const int k = i / N, n = i % N;
  Wt[(size_t)n * ldt + kofs + k] = (bf16_t)W[i];
}

// ---------------------------------------------------------------------------
// Encoder MLP: FIN -> 64 -> 256 -> 128 -> 64. One wave per 16 rows.
// ---------------------------------------------------------------------------
template<int FIN>
__global__ __launch_bounds__(128)
void encoder_mlp_kernel(const float* __restrict__ x0, const float* __restrict__ x1,
                        const float* __restrict__ x2,
                        const float* __restrict__ W1,  const float* __restrict__ b1,
                        const bf16_t* __restrict__ W2t, const float* __restrict__ b2,
                        const bf16_t* __restrict__ W3t, const float* __restrict__ b3,
                        const bf16_t* __restrict__ W4t, const float* __restrict__ b4,
                        bf16_t* __restrict__ out,  int ldo,
                        bf16_t* __restrict__ out2, int ldo2,
                        int nrows) {
  constexpr int LDA = 128 + 8;   // holds h1 (64) then h3 (128)
  constexpr int LDB = 256 + 8;   // holds h2 (256)
  __shared__ __align__(16) bf16_t sA[4][16 * LDA];
  __shared__ __align__(16) bf16_t sB[4][16 * LDB];

  const int lane = threadIdx.x & 31;
  const int wave = threadIdx.x >> 5;
  const int tile = blockIdx.x * 4 + wave;
  if (tile * 16 >= nrows) return;
  const int row0 = tile * 16;
  bf16_t* bufA = sA[wave];
  bf16_t* bufB = sB[wave];

  // L1 (VALU, K tiny): h1 = relu(X @ W1 + b1)  [16,64]
  for (int idx = lane; idx < 16 * 64; idx += 32) {
    const int r = idx >> 6, c = idx & 63;
    float acc = b1[c];
    acc += x0[row0 + r] * W1[0 * 64 + c];
    acc += x1[row0 + r] * W1[1 * 64 + c];
    if (FIN > 2) acc += x2[row0 + r] * W1[2 * 64 + c];
    bufA[r * LDA + c] = (bf16_t)fmaxf(acc, 0.f);
  }
  wave_lds_fence();

  // L2: [16,64] @ [64,256], relu
  #pragma unroll
  for (int t = 0; t < 16; ++t)
    store_tile_lds(wmma_rowtile<64>(bufA, LDA, W2t, t * 16, lane), b2, t * 16, true, bufB, LDB, lane);
  wave_lds_fence();

  // L3: [16,256] @ [256,128], relu
  #pragma unroll
  for (int t = 0; t < 8; ++t)
    store_tile_lds(wmma_rowtile<256>(bufB, LDB, W3t, t * 16, lane), b3, t * 16, true, bufA, LDA, lane);
  wave_lds_fence();

  // L4: [16,128] @ [128,64], no relu, out to global bf16 (optionally twice)
  #pragma unroll
  for (int t = 0; t < 4; ++t) {
    v8f c = wmma_rowtile<128>(bufA, LDA, W4t, t * 16, lane);
    store_tile_glb(c, b4, t * 16, false, out + (size_t)row0 * ldo, ldo, lane);
    if (out2) store_tile_glb(c, b4, t * 16, false, out2 + (size_t)row0 * ldo2, ldo2, lane);
  }
}

// ---------------------------------------------------------------------------
// SAGE aggregation
// ---------------------------------------------------------------------------
__global__ __launch_bounds__(256)
void degree_kernel(const int* __restrict__ dst, float* __restrict__ deg, int nedges) {
  const int i = blockIdx.x * 256 + threadIdx.x;
  if (i < nedges) atomicAdd(&deg[dst[i]], 1.0f);
}

template<int LOGF>
__global__ __launch_bounds__(256)
void sage_scatter_kernel(const bf16_t* __restrict__ h, const bf16_t* __restrict__ e_encs,
                         const int* __restrict__ src, const int* __restrict__ dst,
                         float* __restrict__ agg, int total) {
  const int idx = blockIdx.x * 256 + threadIdx.x;
  if (idx >= total) return;
  const int e = idx >> LOGF;
  const int j = idx & ((1 << LOGF) - 1);
  const float w = (float)e_encs[((size_t)e << 6) + (j & 63)];   // weight broadcast over 64-blocks
  const float m = (float)h[((size_t)src[e] << LOGF) + j] * w;
  atomicAdd(&agg[((size_t)dst[e] << LOGF) + j], m);
}

// out = act([h, agg/max(deg,1)] @ [Wself;Wneigh] + b)
template<int FIN>
__global__ __launch_bounds__(128)
void sage_linear_kernel(const bf16_t* __restrict__ h, int ldh,
                        const float* __restrict__ agg, const float* __restrict__ deg,
                        const bf16_t* __restrict__ Wcat_t,  // [64][2*FIN]
                        const float* __restrict__ bias, int do_relu,
                        bf16_t* __restrict__ out, int ldo, int nrows) {
  constexpr int K   = 2 * FIN;
  constexpr int LDX = K + 8;
  __shared__ __align__(16) bf16_t sX[4][16 * LDX];
  const int lane = threadIdx.x & 31;
  const int wave = threadIdx.x >> 5;
  const int tile = blockIdx.x * 4 + wave;
  if (tile * 16 >= nrows) return;
  const int row0 = tile * 16;
  bf16_t* X = sX[wave];

  // h rows: contiguous bf16 -> async DMA to LDS, 16B chunks (CH per row).
  constexpr int CH = FIN / 8;
  #pragma unroll
  for (int it = 0; it < (16 * CH) / 32; ++it) {
    const int c = it * 32 + lane;
    const int r = c / CH;
    const int q = c % CH;
    async_load_b128(X + r * LDX + q * 8, h + (size_t)(row0 + r) * ldh + q * 8);
  }
  // agg rows: need deg-normalize + f32->bf16, stage through VALU.
  for (int idx = lane; idx < 16 * FIN; idx += 32) {
    const int r = idx / FIN, c = idx % FIN;   // FIN is a power of two -> shifts
    const int n = row0 + r;
    const float d = fmaxf(deg[n], 1.0f);
    X[r * LDX + FIN + c] = (bf16_t)(agg[(size_t)n * FIN + c] / d);
  }
  async_fence();
  wave_lds_fence();

  #pragma unroll
  for (int t = 0; t < 4; ++t)
    store_tile_glb(wmma_rowtile<K>(X, LDX, Wcat_t, t * 16, lane), bias, t * 16,
                   do_relu != 0, out + (size_t)row0 * ldo, ldo, lane);
}

// ---------------------------------------------------------------------------
// Decoder MLP on he = [h3[src], h3[dst]] : 128 -> 64 -> 256 -> 128 -> 1, |.|
// ---------------------------------------------------------------------------
__global__ __launch_bounds__(128)
void decoder_mlp_kernel(const bf16_t* __restrict__ h3,
                        const int* __restrict__ src, const int* __restrict__ dst,
                        const bf16_t* __restrict__ W1t, const float* __restrict__ b1,
                        const bf16_t* __restrict__ W2t, const float* __restrict__ b2,
                        const bf16_t* __restrict__ W3t, const float* __restrict__ b3,
                        const float* __restrict__ W4,   const float* __restrict__ b4,
                        float* __restrict__ P, int nedges) {
  constexpr int LDA = 256 + 8;   // he(128) then h2(256)
  constexpr int LDB = 128 + 8;   // h1(64) then hx(128)
  __shared__ __align__(16) bf16_t sA[4][16 * LDA];
  __shared__ __align__(16) bf16_t sB[4][16 * LDB];
  const int lane = threadIdx.x & 31;
  const int wave = threadIdx.x >> 5;
  const int tile = blockIdx.x * 4 + wave;
  if (tile * 16 >= nedges) return;
  const int e0 = tile * 16;
  bf16_t* bufA = sA[wave];
  bf16_t* bufB = sB[wave];

  // Gather he tile (16 x 128 bf16 = 4KB) straight into LDS: 256 x 16B chunks,
  // 8 async issues per wave. chunk c: row r=c>>4, half hs (src/dst), quad q.
  #pragma unroll
  for (int it = 0; it < 8; ++it) {
    const int c  = it * 32 + lane;
    const int r  = c >> 4;
    const int hs = (c >> 3) & 1;
    const int q  = c & 7;
    const int node = hs ? dst[e0 + r] : src[e0 + r];
    async_load_b128(bufA + r * LDA + hs * 64 + q * 8,
                    h3 + ((size_t)node << 6) + q * 8);
  }
  async_fence();
  wave_lds_fence();

  #pragma unroll
  for (int t = 0; t < 4; ++t)
    store_tile_lds(wmma_rowtile<128>(bufA, LDA, W1t, t * 16, lane), b1, t * 16, true, bufB, LDB, lane);
  wave_lds_fence();
  #pragma unroll
  for (int t = 0; t < 16; ++t)
    store_tile_lds(wmma_rowtile<64>(bufB, LDB, W2t, t * 16, lane), b2, t * 16, true, bufA, LDA, lane);
  wave_lds_fence();
  #pragma unroll
  for (int t = 0; t < 8; ++t)
    store_tile_lds(wmma_rowtile<256>(bufA, LDA, W3t, t * 16, lane), b3, t * 16, true, bufB, LDB, lane);
  wave_lds_fence();

  if (lane < 16) {            // final K=128 -> 1 on VALU
    float s = b4[0];
    #pragma unroll 4
    for (int k = 0; k < 128; ++k)
      s += (float)bufB[lane * LDB + k] * W4[k];
    P[e0 + lane] = fabsf(s);
  }
}

// ---------------------------------------------------------------------------
// Host orchestration
// ---------------------------------------------------------------------------
extern "C" void kernel_launch(void* const* d_in, const int* in_sizes, int n_in,
                              void* d_out, int out_size, void* d_ws, size_t ws_size,
                              hipStream_t stream) {
  const int N = in_sizes[0];      // 50000
  const int E = in_sizes[2];      // 800000

  const float* C   = (const float*)d_in[0];
  const float* Fx  = (const float*)d_in[1];
  const float* A   = (const float*)d_in[2];
  const float* SP1 = (const float*)d_in[3];
  const float* SP0 = (const float*)d_in[4];
  const int*   src = (const int*)d_in[5];
  const int*   dst = (const int*)d_in[6];
  const float *nW1 = (const float*)d_in[7],  *nb1 = (const float*)d_in[8];
  const float *nW2 = (const float*)d_in[9],  *nb2 = (const float*)d_in[10];
  const float *nW3 = (const float*)d_in[11], *nb3 = (const float*)d_in[12];
  const float *nW4 = (const float*)d_in[13], *nb4 = (const float*)d_in[14];
  const float *eW1 = (const float*)d_in[15], *eb1 = (const float*)d_in[16];
  const float *eW2 = (const float*)d_in[17], *eb2 = (const float*)d_in[18];
  const float *eW3 = (const float*)d_in[19], *eb3 = (const float*)d_in[20];
  const float *eW4 = (const float*)d_in[21], *eb4 = (const float*)d_in[22];
  const float *dW1 = (const float*)d_in[23], *db1 = (const float*)d_in[24];
  const float *dW2 = (const float*)d_in[25], *db2 = (const float*)d_in[26];
  const float *dW3 = (const float*)d_in[27], *db3 = (const float*)d_in[28];
  const float *dW4 = (const float*)d_in[29], *db4 = (const float*)d_in[30];
  const float *c1Ws = (const float*)d_in[31], *c1b = (const float*)d_in[32], *c1Wn = (const float*)d_in[33];
  const float *c2Ws = (const float*)d_in[34], *c2b = (const float*)d_in[35], *c2Wn = (const float*)d_in[36];
  float* P = (float*)d_out;

  // ---- workspace bump allocator (total ~154 MB) ----
  char* ws = (char*)d_ws; size_t off = 0;
  auto alloc = [&](size_t bytes) -> void* {
    void* p = ws + off;
    off = (off + bytes + 255) & ~(size_t)255;
    return p;
  };
  bf16_t* nW2t = (bf16_t*)alloc(256 * 64 * 2);
  bf16_t* nW3t = (bf16_t*)alloc(128 * 256 * 2);
  bf16_t* nW4t = (bf16_t*)alloc(64 * 128 * 2);
  bf16_t* eW2t = (bf16_t*)alloc(256 * 64 * 2);
  bf16_t* eW3t = (bf16_t*)alloc(128 * 256 * 2);
  bf16_t* eW4t = (bf16_t*)alloc(64 * 128 * 2);
  bf16_t* dW1t = (bf16_t*)alloc(64 * 128 * 2);
  bf16_t* dW2t = (bf16_t*)alloc(256 * 64 * 2);
  bf16_t* dW3t = (bf16_t*)alloc(128 * 256 * 2);
  bf16_t* cat1 = (bf16_t*)alloc(64 * 128 * 2);
  bf16_t* cat2 = (bf16_t*)alloc(64 * 256 * 2);
  bf16_t* nenc = (bf16_t*)alloc((size_t)N * 64 * 2);
  bf16_t* hA   = (bf16_t*)alloc((size_t)N * 128 * 2);   // [h_conv | n_encs]
  bf16_t* h3   = (bf16_t*)alloc((size_t)N * 64 * 2);
  bf16_t* eenc = (bf16_t*)alloc((size_t)E * 64 * 2);
  float*  deg  = (float*)alloc((size_t)N * 4);
  float*  agg  = (float*)alloc((size_t)N * 128 * 4);
  (void)ws_size; (void)n_in; (void)out_size;

  auto tgrid = [](int elems) { return dim3((elems + 255) / 256); };

  // ---- weight packing (transposed bf16) ----
  transpose_to_bf16_kernel<<<tgrid(64 * 256), 256, 0, stream>>>(nW2, nW2t, 64, 256, 64, 0);
  transpose_to_bf16_kernel<<<tgrid(256 * 128), 256, 0, stream>>>(nW3, nW3t, 256, 128, 256, 0);
  transpose_to_bf16_kernel<<<tgrid(128 * 64), 256, 0, stream>>>(nW4, nW4t, 128, 64, 128, 0);
  transpose_to_bf16_kernel<<<tgrid(64 * 256), 256, 0, stream>>>(eW2, eW2t, 64, 256, 64, 0);
  transpose_to_bf16_kernel<<<tgrid(256 * 128), 256, 0, stream>>>(eW3, eW3t, 256, 128, 256, 0);
  transpose_to_bf16_kernel<<<tgrid(128 * 64), 256, 0, stream>>>(eW4, eW4t, 128, 64, 128, 0);
  transpose_to_bf16_kernel<<<tgrid(128 * 64), 256, 0, stream>>>(dW1, dW1t, 128, 64, 128, 0);
  transpose_to_bf16_kernel<<<tgrid(64 * 256), 256, 0, stream>>>(dW2, dW2t, 64, 256, 64, 0);
  transpose_to_bf16_kernel<<<tgrid(256 * 128), 256, 0, stream>>>(dW3, dW3t, 256, 128, 256, 0);
  transpose_to_bf16_kernel<<<tgrid(64 * 64), 256, 0, stream>>>(c1Ws, cat1, 64, 64, 128, 0);
  transpose_to_bf16_kernel<<<tgrid(64 * 64), 256, 0, stream>>>(c1Wn, cat1, 64, 64, 128, 64);
  transpose_to_bf16_kernel<<<tgrid(128 * 64), 256, 0, stream>>>(c2Ws, cat2, 128, 64, 256, 0);
  transpose_to_bf16_kernel<<<tgrid(128 * 64), 256, 0, stream>>>(c2Wn, cat2, 128, 64, 256, 128);

  // ---- degrees (shared by all three convs) ----
  hipMemsetAsync(deg, 0, (size_t)N * 4, stream);
  degree_kernel<<<tgrid(E), 256, 0, stream>>>(dst, deg, E);

  // ---- encoders ----
  const int nTilesN = N / 16, nTilesE = E / 16;
  encoder_mlp_kernel<2><<<dim3((nTilesN + 3) / 4), 128, 0, stream>>>(
      C, Fx, Fx, nW1, nb1, nW2t, nb2, nW3t, nb3, nW4t, nb4,
      nenc, 64, hA + 64, 128, N);                       // n_encs -> nenc and hA[:,64:]
  encoder_mlp_kernel<3><<<dim3((nTilesE + 3) / 4), 128, 0, stream>>>(
      A, SP1, SP0, eW1, eb1, eW2t, eb2, eW3t, eb3, eW4t, eb4,
      eenc, 64, (bf16_t*)nullptr, 0, E);                // e_encs

  // ---- conv1: fin=64, h=n_encs, relu -> hA[:,0:64] ----
  hipMemsetAsync(agg, 0, (size_t)N * 64 * 4, stream);
  sage_scatter_kernel<6><<<tgrid(E * 64), 256, 0, stream>>>(nenc, eenc, src, dst, agg, E * 64);
  sage_linear_kernel<64><<<dim3((nTilesN + 3) / 4), 128, 0, stream>>>(
      nenc, 64, agg, deg, cat1, c1b, 1, hA, 128, N);

  // ---- conv2 (first): fin=128, h=hA, relu -> hA[:,0:64] ----
  hipMemsetAsync(agg, 0, (size_t)N * 128 * 4, stream);
  sage_scatter_kernel<7><<<tgrid(E * 128), 256, 0, stream>>>(hA, eenc, src, dst, agg, E * 128);
  sage_linear_kernel<128><<<dim3((nTilesN + 3) / 4), 128, 0, stream>>>(
      hA, 128, agg, deg, cat2, c2b, 1, hA, 128, N);

  // ---- conv2 (second, no relu): fin=128 -> h3 ----
  hipMemsetAsync(agg, 0, (size_t)N * 128 * 4, stream);
  sage_scatter_kernel<7><<<tgrid(E * 128), 256, 0, stream>>>(hA, eenc, src, dst, agg, E * 128);
  sage_linear_kernel<128><<<dim3((nTilesN + 3) / 4), 128, 0, stream>>>(
      hA, 128, agg, deg, cat2, c2b, 0, h3, 64, N);

  // ---- decoder ----
  decoder_mlp_kernel<<<dim3((nTilesE + 3) / 4), 128, 0, stream>>>(
      h3, src, dst, dW1t, db1, dW2t, db2, dW3t, db3, dW4, db4, P, E);
}
